// HybridQCLModel_71339406787264
// MI455X (gfx1250) — matrix operand order repaired and checked by reading, not compiled, not run on hardware
//
#include <hip/hip_runtime.h>
#include <math.h>

typedef __attribute__((ext_vector_type(16))) _Float16 v16h;
typedef __attribute__((ext_vector_type(8)))  float    v8f;

#define NQ    8
#define QDEPTH 3
#define FDIM  16
#define HDIM  256
#define PI_F  3.14159265358979f

// ======================================================================
// Kernel 1: fused encoder.
//   h      = relu(x @ w1^T + b1)          (16384x256, K=16)  -> LDS (f16)
//   angles = tanh(h @ w2^T + b2) * pi     (16384x8,  K=256)  -> workspace
// One workgroup = 512 threads = 16 waves handles 16 batch rows.
// Wave w computes the 16x16 h tile for columns [16w,16w+16) with one
// v_wmma_f32_16x16x32_f16 (K padded 16->32). After a barrier, wave 0 chains
// 8 WMMAs over K=256 from LDS to produce the 16x8 angle tile.
// ======================================================================
__global__ __launch_bounds__(512) void encoder_kernel(
    const float* __restrict__ x,  const float* __restrict__ w1,
    const float* __restrict__ b1, const float* __restrict__ w2,
    const float* __restrict__ b2, float* __restrict__ angles)
{
  __shared__ _Float16 hrow[16 * HDIM];          // 16 rows x 256 cols (8 KB)

  const int wave = threadIdx.x >> 5;
  const int lane = threadIdx.x & 31;
  const int hf   = lane >> 4;                   // which 16-lane half
  const int mn   = lane & 15;                   // M (for A) or N (for B/C)
  const int row0 = blockIdx.x * 16;

  // ---------------- GEMM1: 16x16 tile, n-tile = wave ----------------
  {
    // A (16x32 f16): ISA layout -> VGPR j<4: K = 2j + 8*hf (+1);
    // VGPR j>=4 would be K=16..31 -> zero padding (real K = 16).
    v16h a;
    const float* xr = x + (size_t)(row0 + mn) * FDIM + 8 * hf;
    #pragma unroll
    for (int j = 0; j < 4; ++j) {
      a[2*j]   = (_Float16)xr[2*j];
      a[2*j+1] = (_Float16)xr[2*j+1];
    }
    #pragma unroll
    for (int j = 4; j < 8; ++j) { a[2*j] = (_Float16)0.f; a[2*j+1] = (_Float16)0.f; }

    // B (32x16 f16): B[k][n] = w1[n][k]; VGPR j<4: K = 2j + 8*hf (+1);
    // VGPR j>=4 would be K=16..31 (all zero here).
    const int nglob = wave * 16 + mn;
    v16h bm;
    #pragma unroll
    for (int j = 0; j < 8; ++j) {
      float v0 = 0.f, v1 = 0.f;
      if (j < 4) {
        v0 = w1[nglob * FDIM + 8*hf + 2*j];
        v1 = w1[nglob * FDIM + 8*hf + 2*j + 1];
      }
      bm[2*j] = (_Float16)v0; bm[2*j+1] = (_Float16)v1;
    }

    v8f c = {};
    c = __builtin_amdgcn_wmma_f32_16x16x32_f16(false, a, false, bm,
                                               (short)0, c, false, false);

    // C layout: VGPR i -> (M = i + 8*hf, N = mn). Bias + ReLU -> LDS f16.
    const float bias = b1[nglob];
    #pragma unroll
    for (int i = 0; i < 8; ++i) {
      const int m = i + 8 * hf;
      float v = c[i] + bias;
      v = v > 0.f ? v : 0.f;
      hrow[m * HDIM + nglob] = (_Float16)v;
    }
  }
  __syncthreads();

  // ---------------- GEMM2 (wave 0, full EXEC): 16x8 angles ----------------
  if (wave == 0) {
    v8f acc = {};
    #pragma unroll
    for (int kc = 0; kc < 8; ++kc) {
      v16h a;
      #pragma unroll
      for (int j = 0; j < 8; ++j) {
        const int k = kc * 32 + ((j < 4) ? (2*j + 8*hf) : (16 + 2*(j-4) + 8*hf));
        a[2*j]   = hrow[mn * HDIM + k];
        a[2*j+1] = hrow[mn * HDIM + k + 1];
      }
      v16h bm;
      #pragma unroll
      for (int j = 0; j < 8; ++j) {
        const int k = kc * 32 + ((j < 4) ? (2*j + 8*hf) : (16 + 2*(j-4) + 8*hf));
        float v0 = 0.f, v1 = 0.f;
        if (mn < NQ) {                      // only 8 real output columns
          v0 = w2[mn * HDIM + k];
          v1 = w2[mn * HDIM + k + 1];
        }
        bm[2*j] = (_Float16)v0; bm[2*j+1] = (_Float16)v1;
      }
      acc = __builtin_amdgcn_wmma_f32_16x16x32_f16(false, a, false, bm,
                                                   (short)0, acc, false, false);
    }
    if (mn < NQ) {
      const float bias = b2[mn];
      #pragma unroll
      for (int i = 0; i < 8; ++i) {
        const int m = i + 8 * hf;
        angles[(size_t)(row0 + m) * NQ + mn] = tanhf(acc[i] + bias) * PI_F;
      }
    }
  }
}

// ======================================================================
// Kernel 2: 8-qubit statevector sim, one wave32 per batch element.
// Amplitude index i in [0,256): lane = i & 31, register r = i >> 5.
// Reference qubit q (MSB-first) acts on bit p = 7 - q of i:
//   p in {5,6,7} -> register-index bit (pure VALU pairing)
//   p in {0..4}  -> lane bit (cross-lane via __shfl_xor -> ds permute)
// Fused: expval butterfly-reduce -> decoder dot -> sigmoid -> out.
// ======================================================================
__device__ __forceinline__ void ry_gate(float* sr, float* si, float th,
                                        int p, int lane)
{
  float s, c;
  __sincosf(0.5f * th, &s, &c);
  if (p >= 5) {
    const int d = 1 << (p - 5);
    #pragma unroll
    for (int r = 0; r < 8; ++r) {
      if ((r & d) == 0) {
        const int r1 = r | d;
        float a0r = sr[r], a0i = si[r], a1r = sr[r1], a1i = si[r1];
        sr[r]  = c * a0r - s * a1r;  si[r]  = c * a0i - s * a1i;
        sr[r1] = s * a0r + c * a1r;  si[r1] = s * a0i + c * a1i;
      }
    }
  } else {
    const int m = 1 << p;
    // bit==0: new = c*mine - s*partner ; bit==1: new = c*mine + s*partner
    const float sl = ((lane & m) ? s : -s);
    #pragma unroll
    for (int r = 0; r < 8; ++r) {
      float pr = __shfl_xor(sr[r], m, 32);
      float pi = __shfl_xor(si[r], m, 32);
      sr[r] = fmaf(sl, pr, c * sr[r]);
      si[r] = fmaf(sl, pi, c * si[r]);
    }
  }
}

__device__ __forceinline__ void rz_gate(float* sr, float* si, float th,
                                        int p, int lane)
{
  float s, c;
  __sincosf(0.5f * th, &s, &c);
  if (p >= 5) {
    #pragma unroll
    for (int r = 0; r < 8; ++r) {
      const float sg = ((r >> (p - 5)) & 1) ? s : -s;   // bit0 -> e^{-ih}
      const float nr = c * sr[r] - sg * si[r];
      const float ni = c * si[r] + sg * sr[r];
      sr[r] = nr; si[r] = ni;
    }
  } else {
    const float sg = ((lane >> p) & 1) ? s : -s;
    #pragma unroll
    for (int r = 0; r < 8; ++r) {
      const float nr = c * sr[r] - sg * si[r];
      const float ni = c * si[r] + sg * sr[r];
      sr[r] = nr; si[r] = ni;
    }
  }
}

__device__ __forceinline__ void cx_gate(float* sr, float* si,
                                        int pc, int pt, int lane)
{
  if (pt >= 5) {
    const int d = 1 << (pt - 5);
    if (pc >= 5) {
      #pragma unroll
      for (int r = 0; r < 8; ++r) {
        if (((r & d) == 0) && (((r >> (pc - 5)) & 1) != 0)) {
          const int r1 = r | d;
          float t = sr[r]; sr[r] = sr[r1]; sr[r1] = t;
          t = si[r]; si[r] = si[r1]; si[r1] = t;
        }
      }
    } else {
      const bool ctl = ((lane >> pc) & 1) != 0;
      #pragma unroll
      for (int r = 0; r < 8; ++r) {
        if ((r & d) == 0) {
          const int r1 = r | d;
          float t0 = sr[r], t1 = sr[r1];
          sr[r] = ctl ? t1 : t0;  sr[r1] = ctl ? t0 : t1;
          t0 = si[r]; t1 = si[r1];
          si[r] = ctl ? t1 : t0;  si[r1] = ctl ? t0 : t1;
        }
      }
    }
  } else {
    const int m = 1 << pt;
    #pragma unroll
    for (int r = 0; r < 8; ++r) {
      const bool ctl = (pc >= 5) ? (((r >> (pc - 5)) & 1) != 0)
                                 : (((lane >> pc) & 1) != 0);
      float pr = __shfl_xor(sr[r], m, 32);
      float pi = __shfl_xor(si[r], m, 32);
      sr[r] = ctl ? pr : sr[r];
      si[r] = ctl ? pi : si[r];
    }
  }
}

__global__ __launch_bounds__(256) void qsim_decoder_kernel(
    const float* __restrict__ angles, const float* __restrict__ qp,
    const float* __restrict__ dw1, const float* __restrict__ db1,
    const float* __restrict__ dw2, const float* __restrict__ db2,
    float* __restrict__ out)
{
  const int lane  = threadIdx.x & 31;
  const int batch = blockIdx.x * 8 + (threadIdx.x >> 5);

  float sr[8], si[8];
  #pragma unroll
  for (int r = 0; r < 8; ++r) { sr[r] = 0.f; si[r] = 0.f; }
  if (lane == 0) sr[0] = 1.f;                    // |00000000>

  const float* ang = angles + (size_t)batch * NQ;
  #pragma unroll
  for (int q = 0; q < NQ; ++q) ry_gate(sr, si, ang[q], 7 - q, lane);

  int idx = 0;
  #pragma unroll
  for (int d = 0; d < QDEPTH; ++d) {
    #pragma unroll
    for (int q = 0; q < NQ; ++q) {
      rz_gate(sr, si, qp[idx],     7 - q, lane);
      ry_gate(sr, si, qp[idx + 1], 7 - q, lane);
      idx += 2;
    }
    #pragma unroll
    for (int i = 0; i < NQ; ++i)
      cx_gate(sr, si, 7 - i, 7 - ((i + 1) & 7), lane);
  }

  // <Z> on qubit 0 (MSB = bit 7 of i = bit 2 of r)
  float e = 0.f;
  #pragma unroll
  for (int r = 0; r < 8; ++r) {
    const float p = sr[r] * sr[r] + si[r] * si[r];
    e += (r & 4) ? -p : p;
  }
  #pragma unroll
  for (int off = 16; off >= 1; off >>= 1) e += __shfl_xor(e, off, 32);

  // Fused decoder: sigmoid( sum_j relu(e*dw1[j]+db1[j]) * dw2[j] + db2 )
  float acc = 0.f;
  #pragma unroll
  for (int t = 0; t < 8; ++t) {
    const int j = lane + 32 * t;
    float v = fmaf(e, dw1[j], db1[j]);
    v = v > 0.f ? v : 0.f;
    acc += v * dw2[j];
  }
  #pragma unroll
  for (int off = 16; off >= 1; off >>= 1) acc += __shfl_xor(acc, off, 32);

  if (lane == 0)
    out[batch] = 1.f / (1.f + __expf(-(acc + db2[0])));
}

// ======================================================================
extern "C" void kernel_launch(void* const* d_in, const int* in_sizes, int n_in,
                              void* d_out, int out_size, void* d_ws, size_t ws_size,
                              hipStream_t stream)
{
  const float* x   = (const float*)d_in[0];
  const float* w1  = (const float*)d_in[1];
  const float* b1  = (const float*)d_in[2];
  const float* w2  = (const float*)d_in[3];
  const float* b2  = (const float*)d_in[4];
  const float* qp  = (const float*)d_in[5];
  const float* dw1 = (const float*)d_in[6];
  const float* db1 = (const float*)d_in[7];
  const float* dw2 = (const float*)d_in[8];
  const float* db2 = (const float*)d_in[9];

  float* out    = (float*)d_out;
  float* angles = (float*)d_ws;                 // B*8 floats (512 KB @ B=16K)

  const int B = in_sizes[0] / FDIM;             // 16384
  if (B <= 0) return;

  encoder_kernel<<<B / 16, 512, 0, stream>>>(x, w1, b1, w2, b2, angles);
  qsim_decoder_kernel<<<B / 8, 256, 0, stream>>>(angles, qp, dw1, db1, dw2, db2, out);
}
